// MaskedAutoencoderGroupChannelViT_25400436589034
// MI455X (gfx1250) — compile-verified, matrix operand order checked by reading, not run on hardware
//
#include <hip/hip_runtime.h>

#define B_ 32
#define G_ 3
#define L_ 196
#define D_ 1024
#define SEQ_ (1 + G_ * L_)          // 589
#define ROWS_ (B_ * G_ * L_)        // 18816
#define PI_F 3.14159265358979323846f

typedef __attribute__((ext_vector_type(16))) __bf16 v16bf;
typedef __attribute__((ext_vector_type(4)))  __bf16 v4bf;
typedef __attribute__((ext_vector_type(8)))  float  v8f;
typedef __attribute__((ext_vector_type(4)))  float  v4f;
typedef __attribute__((ext_vector_type(2)))  float  v2f;
typedef __attribute__((ext_vector_type(4)))  unsigned int v4u;
typedef __attribute__((ext_vector_type(8)))  int v8i;
typedef __attribute__((ext_vector_type(4)))  int v4i;

__device__ __forceinline__ v8f vzero8() {
    v8f z = {0.f, 0.f, 0.f, 0.f, 0.f, 0.f, 0.f, 0.f};
    return z;
}

__device__ __forceinline__ v8f wmma_bf16(v16bf a, v16bf b, v8f c) {
    return __builtin_amdgcn_wmma_f32_16x16x32_bf16(false, a, false, b, (short)0, c, false, false);
}

__device__ __forceinline__ v8f wmma_f32(v2f a, v2f b, v8f c) {
    return __builtin_amdgcn_wmma_f32_16x16x4_f32(false, a, false, b, (short)0, c, false, false);
}

// ---- Tensor Data Mover: 2-D tile of bf16 (rows x tile_k, row stride in elems) -> LDS
__device__ __forceinline__ void tdm_load_2d(unsigned int lds_off, const void* gsrc,
                                            unsigned int tile_k, unsigned int rows,
                                            unsigned int stride_elems) {
    unsigned long long ga = (unsigned long long)(uintptr_t)gsrc;
    v4u g0 = { 1u,                                   // count=1, user descriptor
               lds_off,                              // lds_addr (bytes)
               (unsigned int)ga,                     // global_addr[31:0]
               (unsigned int)((ga >> 32) & 0x1FFFFFFull) | (2u << 30) };  // ga[56:32], type=2
    v8i g1 = { (int)0x00010000,                                            // data_size=1 (2B)
               (int)((tile_k & 0xFFFFu) << 16),                            // tensor_dim0[15:0]
               (int)(((tile_k >> 16) & 0xFFFFu) | ((rows & 0xFFFFu) << 16)),// dim0 hi / dim1 lo
               (int)(((rows >> 16) & 0xFFFFu) | ((tile_k & 0xFFFFu) << 16)),// dim1 hi / tile_dim0
               (int)(rows & 0xFFFFu),                                      // tile_dim1, tile_dim2=0
               (int)stride_elems,                                          // tensor_dim0_stride lo
               0, 0 };
    v4i g2 = {0, 0, 0, 0};
    v4i g3 = {0, 0, 0, 0};
#if defined(__clang_major__) && (__clang_major__ >= 23)
    v8i g4 = {0, 0, 0, 0, 0, 0, 0, 0};
    __builtin_amdgcn_tensor_load_to_lds(g0, g1, g2, g3, g4, 0);
#else
    __builtin_amdgcn_tensor_load_to_lds(g0, g1, g2, g3, 0);
#endif
}

// ---------------------------------------------------------------- weight prep
__global__ void k_cvt_transpose(const float* __restrict__ W, __bf16* __restrict__ Wt,
                                int K, int N) {
    int i = blockIdx.x * blockDim.x + threadIdx.x;
    if (i < K * N) {
        int k = i / N, n = i % N;
        Wt[(size_t)n * K + k] = (__bf16)W[i];
    }
}

__global__ void k_cvt_conv(const float* __restrict__ W, __bf16* __restrict__ Wt) {
    // W: (O=1024, I=1024, H=3, 1) -> Wt[h][o][i]
    int i = blockIdx.x * blockDim.x + threadIdx.x;
    if (i < D_ * D_ * 3) {
        int h = i % 3;
        int rem = i / 3;
        int in = rem % D_;
        int o  = rem / D_;
        Wt[((size_t)h * D_ + o) * D_ + in] = (__bf16)W[i];
    }
}

__global__ void k_twiddle(float* __restrict__ twC, float* __restrict__ twS) {
    int i = blockIdx.x * blockDim.x + threadIdx.x;
    if (i < L_ * L_) {
        int l = i / L_, k = i % L_;
        float ang = 2.0f * PI_F * (float)((l * k) % L_) / (float)L_;
        twC[i] = cosf(ang);
        twS[i] = -sinf(ang);
    }
}

// ---------------------------------------------------------------- gf = mean over G (vec4)
__global__ void k_gf(const float* __restrict__ x, float* __restrict__ gf) {
    int i = blockIdx.x * blockDim.x + threadIdx.x;
    if (i < B_ * L_ * D_ / 4) {
        int d4  = i & (D_ / 4 - 1);
        int rem = i / (D_ / 4);
        int l = rem % L_;
        int b = rem / L_;
        size_t base = ((size_t)b * SEQ_ + 1 + l) * D_ + 4 * d4;
        v4f a = *(const v4f*)(x + base);
        v4f c = *(const v4f*)(x + base + (size_t)L_ * D_);
        v4f e = *(const v4f*)(x + base + 2ull * L_ * D_);
        v4f r = { (a.x + c.x + e.x) * (1.f / 3.f), (a.y + c.y + e.y) * (1.f / 3.f),
                  (a.z + c.z + e.z) * (1.f / 3.f), (a.w + c.w + e.w) * (1.f / 3.f) };
        *(v4f*)(gf + (size_t)i * 4) = r;
    }
}

// ---------------------------------------------------------------- gate MLP + blend
// A double-buffered in LDS; K split into seq-half then gf-half.
__global__ void k_gate_xg(const float* __restrict__ x, const float* __restrict__ gf,
                          const __bf16* __restrict__ w1t, const float* __restrict__ b1,
                          const float* __restrict__ lnw, const float* __restrict__ lnb,
                          const float* __restrict__ w2, const float* __restrict__ b2,
                          __bf16* __restrict__ xg1b) {
    __shared__ __align__(32) __bf16 Alds[2][16][32];
    __shared__ float H[16][512];
    __shared__ float red[16][16];
    __shared__ float smean[16], srstd[16], sgate[16];
    __shared__ size_t soff[16], goff[16];

    const int tid = threadIdx.x;
    const int lane = tid & 31, wave = tid >> 5;
    const int rbase = blockIdx.x * 16;

    if (tid < 16) {
        int r = rbase + tid;
        int b = r / (G_ * L_);
        int rem = r % (G_ * L_);
        int g = rem / L_, l = rem % L_;
        soff[tid] = ((size_t)b * SEQ_ + 1 + (size_t)g * L_ + l) * D_;
        goff[tid] = ((size_t)b * L_ + l) * D_;
    }
    __syncthreads();

    const int sm = (tid & 127) >> 3, sc = (tid & 7) << 2;
    const float* ps = x + soff[sm] + sc;     // seq half source
    const float* pg = gf + goff[sm] + sc;    // gf half source

    auto stageA = [&](int it, int buf) {
        if (tid < 128) {
            const float* src = ((it < 32) ? ps : pg) + ((it & 31) << 5);
            v4f v = *(const v4f*)src;
            v4bf bv = { (__bf16)v.x, (__bf16)v.y, (__bf16)v.z, (__bf16)v.w };
            *(v4bf*)&Alds[buf][sm][sc] = bv;
        }
    };

    v8f acc[4];
    #pragma unroll
    for (int j = 0; j < 4; ++j) acc[j] = vzero8();

    stageA(0, 0);
    __syncthreads();
    for (int it = 0; it < 64; ++it) {
        const int buf = it & 1;
        if (it + 1 < 64) stageA(it + 1, buf ^ 1);
        v16bf a = *(const v16bf*)&Alds[buf][lane & 15][(lane >> 4) * 16];
        const int kcol = it << 5;
        #pragma unroll
        for (int j = 0; j < 4; ++j) {
            int ncol = (wave * 4 + j) * 16 + (lane & 15);
            const __bf16* bp = w1t + (size_t)ncol * 2048 + kcol + (lane >> 4) * 16;
            __builtin_prefetch((const void*)(bp + 32), 0, 1);
            acc[j] = wmma_bf16(a, *(const v16bf*)bp, acc[j]);
        }
        __syncthreads();
    }

    #pragma unroll
    for (int j = 0; j < 4; ++j) {
        int ncol = (wave * 4 + j) * 16 + (lane & 15);
        float bias = b1[ncol];
        #pragma unroll
        for (int t = 0; t < 8; ++t)
            H[t + 8 * (lane >> 4)][ncol] = acc[j][t] + bias;
    }
    __syncthreads();

    const int row = tid >> 4, c0 = tid & 15;
    float s = 0.f;
    for (int c = c0; c < 512; c += 16) s += H[row][c];
    red[row][c0] = s;
    __syncthreads();
    if (c0 == 0) {
        float t = 0.f;
        for (int i = 0; i < 16; ++i) t += red[row][i];
        smean[row] = t / 512.f;
    }
    __syncthreads();
    float mu = smean[row];
    float vv = 0.f;
    for (int c = c0; c < 512; c += 16) { float d = H[row][c] - mu; vv += d * d; }
    red[row][c0] = vv;
    __syncthreads();
    if (c0 == 0) {
        float t = 0.f;
        for (int i = 0; i < 16; ++i) t += red[row][i];
        srstd[row] = rsqrtf(t / 512.f + 1e-5f);
    }
    __syncthreads();
    float rs = srstd[row];
    float p = 0.f;
    for (int c = c0; c < 512; c += 16) {
        float h = (H[row][c] - mu) * rs * lnw[c] + lnb[c];
        h = h > 0.f ? h : 0.f;
        p += h * w2[c];
    }
    red[row][c0] = p;
    __syncthreads();
    if (c0 == 0) {
        float t = 0.f;
        for (int i = 0; i < 16; ++i) t += red[row][i];
        t += b2[0];
        sgate[row] = 1.f / (1.f + expf(-t));
    }
    __syncthreads();

    for (int e4 = tid; e4 < 4 * D_; e4 += 256) {          // 4096 groups of 4 elems
        int m = e4 >> 8;
        int d = (e4 & 255) << 2;
        float gate = sgate[m];
        v4f sv = *(const v4f*)(x + soff[m] + d);
        v4f fv = *(const v4f*)(gf + goff[m] + d);
        v4bf o = { (__bf16)(sv.x * gate + fv.x * (1.f - gate)),
                   (__bf16)(sv.y * gate + fv.y * (1.f - gate)),
                   (__bf16)(sv.z * gate + fv.z * (1.f - gate)),
                   (__bf16)(sv.w * gate + fv.w * (1.f - gate)) };
        *(v4bf*)(xg1b + ((size_t)(rbase + m)) * D_ + d) = o;
    }
}

// ---------------------------------------------------------------- conv over G (K=3*1024)
// Double-buffered: TDM DMA of next B panel + A ds-stores overlap current WMMAs.
__global__ void k_conv(const __bf16* __restrict__ xg1b, const __bf16* __restrict__ wt,
                       const float* __restrict__ bias, float* __restrict__ xa) {
    __shared__ __align__(32)  __bf16 Alds[2][16][32];
    __shared__ __align__(128) __bf16 Blds[2][256 * 32];     // 2 x 16 KB panels
    __shared__ long long soff[3][16];

    const int tid = threadIdx.x, lane = tid & 31, wave = tid >> 5;
    const int rbase = blockIdx.x * 16;
    const int colbase = blockIdx.y * 256;

    if (tid < 48) {
        int dg = tid / 16, m = tid % 16;
        int r = rbase + m;
        int b = r / (G_ * L_);
        int rem = r % (G_ * L_);
        int g = rem / L_, l = rem % L_;
        int g2 = g + dg - 1;
        soff[dg][m] = (g2 < 0 || g2 >= G_) ? -1ll
                      : (long long)((((size_t)b * G_ + g2) * L_ + l) * D_);
    }
    __syncthreads();

    const int sm = (tid & 127) >> 3, sc = (tid & 7) << 2;
    long long o0 = soff[0][sm], o1 = soff[1][sm], o2 = soff[2][sm];
    const bool av0 = o0 >= 0, av1 = o1 >= 0, av2 = o2 >= 0;
    const __bf16* ap0 = xg1b + (av0 ? o0 : 0) + sc;
    const __bf16* ap1 = xg1b + (av1 ? o1 : 0) + sc;
    const __bf16* ap2 = xg1b + (av2 ? o2 : 0) + sc;

    const unsigned int b0off = (unsigned int)(unsigned long long)(uintptr_t)&Blds[0][0];
    const unsigned int b1off = (unsigned int)(unsigned long long)(uintptr_t)&Blds[1][0];

    auto stageC = [&](int it, int buf) {
        const int dg = it >> 5, kc = (it & 31) << 5;
        if (wave == 0) {
            const __bf16* gsrc = wt + ((size_t)dg * D_ + colbase) * D_ + kc;
            tdm_load_2d(buf ? b1off : b0off, gsrc, 32u, 256u, (unsigned int)D_);
        }
        if (tid < 128) {
            const __bf16* ap = (dg == 0) ? ap0 : (dg == 1) ? ap1 : ap2;
            const bool av   = (dg == 0) ? av0 : (dg == 1) ? av1 : av2;
            v4bf v = { (__bf16)0.f, (__bf16)0.f, (__bf16)0.f, (__bf16)0.f };
            if (av) v = *(const v4bf*)(ap + kc);
            *(v4bf*)&Alds[buf][sm][sc] = v;
        }
    };

    v8f acc[2];
    acc[0] = vzero8();
    acc[1] = vzero8();

    stageC(0, 0);
    if (wave == 0) __builtin_amdgcn_s_wait_tensorcnt(0);
    __syncthreads();

    for (int it = 0; it < 96; ++it) {
        const int buf = it & 1;
        if (it + 1 < 96) stageC(it + 1, buf ^ 1);          // DMA overlaps WMMAs below
        v16bf a = *(const v16bf*)&Alds[buf][lane & 15][(lane >> 4) * 16];
        #pragma unroll
        for (int j = 0; j < 2; ++j) {
            int nloc = (wave * 2 + j) * 16 + (lane & 15);
            v16bf b = *(const v16bf*)&Blds[buf][nloc * 32 + (lane >> 4) * 16];
            acc[j] = wmma_bf16(a, b, acc[j]);
        }
        if (wave == 0) __builtin_amdgcn_s_wait_tensorcnt(0);
        __syncthreads();
    }

    #pragma unroll
    for (int j = 0; j < 2; ++j) {
        int ncol = colbase + (wave * 2 + j) * 16 + (lane & 15);
        float bv = bias[ncol];
        #pragma unroll
        for (int t = 0; t < 8; ++t) {
            int row = t + 8 * (lane >> 4);
            xa[((size_t)(rbase + row)) * D_ + ncol] = acc[j][t] + bv;
        }
    }
}

// ---------------------------------------------------------------- forward DFT as f32 GEMM
__global__ void k_dft(const float* __restrict__ xa, const float* __restrict__ twC,
                      const float* __restrict__ twS, float* __restrict__ reb,
                      float* __restrict__ imb) {
    const int tid = threadIdx.x, lane = tid & 31, wave = tid >> 5;
    const int bg = blockIdx.x, mtile = blockIdx.y;
    const int col = blockIdx.z * 128 + wave * 16 + (lane & 15);
    const int kk = (lane >> 4) * 2;
    const int m = mtile * 16 + (lane & 15);
    const float* X = xa + (size_t)bg * L_ * D_;

    v8f aR = vzero8(), aI = vzero8();
    for (int k = 0; k < L_; k += 4) {
        v2f ac = {0.f, 0.f}, as = {0.f, 0.f}, bb;
        if (m < L_) {
            ac = *(const v2f*)&twC[m * L_ + k + kk];
            as = *(const v2f*)&twS[m * L_ + k + kk];
        }
        bb.x = X[(size_t)(k + kk) * D_ + col];
        bb.y = X[(size_t)(k + kk + 1) * D_ + col];
        aR = wmma_f32(ac, bb, aR);
        aI = wmma_f32(as, bb, aI);
    }
    #pragma unroll
    for (int t = 0; t < 8; ++t) {
        int row = mtile * 16 + t + 8 * (lane >> 4);
        if (row < L_) {
            size_t o = ((size_t)bg * L_ + row) * D_ + col;
            reb[o] = aR[t];
            imb[o] = aI[t];
        }
    }
}

// ---------------------------------------------------------------- mag/phase (f32 in place + bf16 copies)
__global__ void k_magph(float* __restrict__ reb, float* __restrict__ imb,
                        __bf16* __restrict__ magb, __bf16* __restrict__ phb, size_t n4) {
    size_t i = (size_t)blockIdx.x * blockDim.x + threadIdx.x;
    if (i < n4) {
        v4f re = *(const v4f*)(reb + 4 * i);
        v4f im = *(const v4f*)(imb + 4 * i);
        v4f mg, ph;
        mg.x = sqrtf(re.x * re.x + im.x * im.x); ph.x = atan2f(im.x, re.x);
        mg.y = sqrtf(re.y * re.y + im.y * im.y); ph.y = atan2f(im.y, re.y);
        mg.z = sqrtf(re.z * re.z + im.z * im.z); ph.z = atan2f(im.z, re.z);
        mg.w = sqrtf(re.w * re.w + im.w * im.w); ph.w = atan2f(im.w, re.w);
        *(v4f*)(reb + 4 * i) = mg;
        *(v4f*)(imb + 4 * i) = ph;
        v4bf mb = { (__bf16)mg.x, (__bf16)mg.y, (__bf16)mg.z, (__bf16)mg.w };
        v4bf pb = { (__bf16)ph.x, (__bf16)ph.y, (__bf16)ph.z, (__bf16)ph.w };
        *(v4bf*)(magb + 4 * i) = mb;
        *(v4bf*)(phb + 4 * i) = pb;
    }
}

// ---------------------------------------------------------------- attention MLP 1024->256->1024
// Phase-1 weight panel via double-buffered TDM; phase-2 streams weights with clauses.
__global__ void k_mlp_att(const __bf16* __restrict__ X, const __bf16* __restrict__ w1t,
                          const float* __restrict__ b1, const float* __restrict__ lnw,
                          const float* __restrict__ lnb, const __bf16* __restrict__ w2t,
                          const float* __restrict__ b2, int act, float* __restrict__ out) {
    __shared__ __align__(32)  __bf16 Alds[2][16][32];
    __shared__ __align__(128) __bf16 Blds[2][256 * 32];     // 2 x 16 KB
    __shared__ float H[16][256];
    __shared__ __align__(32) __bf16 Hb[16][256];
    __shared__ float red[16][16];
    __shared__ float smean[16], srstd[16];

    const int tid = threadIdx.x, lane = tid & 31, wave = tid >> 5;
    const size_t rowbase = (size_t)blockIdx.x * 16;
    const __bf16* Xr = X + rowbase * D_;

    const int sm = (tid & 127) >> 3, sc = (tid & 7) << 2;
    const __bf16* pA = Xr + (size_t)sm * D_ + sc;

    const unsigned int b0off = (unsigned int)(unsigned long long)(uintptr_t)&Blds[0][0];
    const unsigned int b1off = (unsigned int)(unsigned long long)(uintptr_t)&Blds[1][0];

    auto stage1 = [&](int it, int buf) {
        const int kc = it << 5;
        if (wave == 0)
            tdm_load_2d(buf ? b1off : b0off, w1t + kc, 32u, 256u, (unsigned int)D_);
        if (tid < 128)
            *(v4bf*)&Alds[buf][sm][sc] = *(const v4bf*)(pA + kc);
    };

    v8f acc1[2];
    acc1[0] = vzero8();
    acc1[1] = vzero8();

    stage1(0, 0);
    if (wave == 0) __builtin_amdgcn_s_wait_tensorcnt(0);
    __syncthreads();

    for (int it = 0; it < 32; ++it) {
        const int buf = it & 1;
        if (it + 1 < 32) stage1(it + 1, buf ^ 1);
        v16bf a = *(const v16bf*)&Alds[buf][lane & 15][(lane >> 4) * 16];
        #pragma unroll
        for (int j = 0; j < 2; ++j) {
            int nloc = (wave * 2 + j) * 16 + (lane & 15);
            v16bf b = *(const v16bf*)&Blds[buf][nloc * 32 + (lane >> 4) * 16];
            acc1[j] = wmma_bf16(a, b, acc1[j]);
        }
        if (wave == 0) __builtin_amdgcn_s_wait_tensorcnt(0);
        __syncthreads();
    }

    #pragma unroll
    for (int j = 0; j < 2; ++j) {
        int ncol = (wave * 2 + j) * 16 + (lane & 15);
        float bias = b1[ncol];
        #pragma unroll
        for (int t = 0; t < 8; ++t)
            H[t + 8 * (lane >> 4)][ncol] = acc1[j][t] + bias;
    }
    __syncthreads();

    const int row = tid >> 4, c0 = tid & 15;
    float s = 0.f;
    for (int c = c0; c < 256; c += 16) s += H[row][c];
    red[row][c0] = s;
    __syncthreads();
    if (c0 == 0) {
        float t = 0.f;
        for (int i = 0; i < 16; ++i) t += red[row][i];
        smean[row] = t / 256.f;
    }
    __syncthreads();
    float mu = smean[row];
    float vv = 0.f;
    for (int c = c0; c < 256; c += 16) { float d = H[row][c] - mu; vv += d * d; }
    red[row][c0] = vv;
    __syncthreads();
    if (c0 == 0) {
        float t = 0.f;
        for (int i = 0; i < 16; ++i) t += red[row][i];
        srstd[row] = rsqrtf(t / 256.f + 1e-5f);
    }
    __syncthreads();
    float rs = srstd[row];
    for (int c = c0; c < 256; c += 16) {
        float h = (H[row][c] - mu) * rs * lnw[c] + lnb[c];
        Hb[row][c] = (__bf16)(h > 0.f ? h : 0.f);
    }
    __syncthreads();

    v8f acc2[8];
    #pragma unroll
    for (int j = 0; j < 8; ++j) acc2[j] = vzero8();
    for (int kc = 0; kc < 256; kc += 32) {
        v16bf a = *(const v16bf*)&Hb[lane & 15][kc + (lane >> 4) * 16];
        #pragma unroll
        for (int j = 0; j < 8; ++j) {
            int ncol = (wave * 8 + j) * 16 + (lane & 15);
            const __bf16* bp = w2t + (size_t)ncol * 256 + kc + (lane >> 4) * 16;
            acc2[j] = wmma_bf16(a, *(const v16bf*)bp, acc2[j]);
        }
    }
    #pragma unroll
    for (int j = 0; j < 8; ++j) {
        int ncol = (wave * 8 + j) * 16 + (lane & 15);
        float bias = b2[ncol];
        #pragma unroll
        for (int t = 0; t < 8; ++t) {
            int rr = t + 8 * (lane >> 4);
            float val = acc2[j][t] + bias;
            val = (act == 0) ? (1.f / (1.f + expf(-val))) : tanhf(val);
            out[(rowbase + rr) * D_ + ncol] = val;
        }
    }
}

// ---------------------------------------------------------------- band MLP + freq weights
__global__ void k_band(const float* __restrict__ xa, const float* __restrict__ w1,
                       const float* __restrict__ b1, const float* __restrict__ lnw,
                       const float* __restrict__ lnb, const float* __restrict__ w2,
                       const float* __restrict__ b2, float* __restrict__ wband) {
    __shared__ float meanv[1024];
    __shared__ float hidv[256];
    __shared__ float red[256];
    __shared__ float bw[3];
    __shared__ float mu_s, rs_s;

    const int tid = threadIdx.x;
    const float* Xb = xa + (size_t)blockIdx.x * L_ * D_;

    for (int c = tid; c < D_; c += 256) {
        float s = 0.f;
        for (int l = 0; l < L_; ++l) s += Xb[(size_t)l * D_ + c];
        meanv[c] = s / (float)L_;
    }
    __syncthreads();
    {
        float h = b1[tid];
        for (int k = 0; k < D_; ++k) h += meanv[k] * w1[(size_t)k * 256 + tid];
        hidv[tid] = h;
    }
    __syncthreads();
    red[tid] = hidv[tid];
    __syncthreads();
    for (int s = 128; s > 0; s >>= 1) {
        if (tid < s) red[tid] += red[tid + s];
        __syncthreads();
    }
    if (tid == 0) mu_s = red[0] / 256.f;
    __syncthreads();
    float mu = mu_s;
    float d = hidv[tid] - mu;
    red[tid] = d * d;
    __syncthreads();
    for (int s = 128; s > 0; s >>= 1) {
        if (tid < s) red[tid] += red[tid + s];
        __syncthreads();
    }
    if (tid == 0) rs_s = rsqrtf(red[0] / 256.f + 1e-5f);
    __syncthreads();
    float h = (hidv[tid] - mu) * rs_s * lnw[tid] + lnb[tid];
    hidv[tid] = h > 0.f ? h : 0.f;
    __syncthreads();
    if (tid < 3) {
        float o = b2[tid];
        for (int c = 0; c < 256; ++c) o += hidv[c] * w2[(size_t)c * 3 + tid];
        bw[tid] = o;
    }
    __syncthreads();
    if (tid == 0) {
        float mx = fmaxf(bw[0], fmaxf(bw[1], bw[2]));
        float e0 = expf(bw[0] - mx), e1 = expf(bw[1] - mx), e2 = expf(bw[2] - mx);
        float inv = 1.f / (e0 + e1 + e2);
        bw[0] = e0 * inv; bw[1] = e1 * inv; bw[2] = e2 * inv;
    }
    __syncthreads();
    if (tid < L_) {
        int f = (tid <= L_ / 2) ? tid : (L_ - tid);   // |fftfreq * L|
        int idx = (f <= L_ / 8) ? 0 : ((f <= L_ / 4) ? 1 : 2);
        wband[(size_t)blockIdx.x * L_ + tid] = bw[idx];
    }
}

// ---------------------------------------------------------------- combine -> complex
__global__ void k_combine(float* __restrict__ reb, float* __restrict__ imb,
                          const float* __restrict__ matt, const float* __restrict__ patt,
                          const float* __restrict__ wband, size_t n) {
    size_t i = (size_t)blockIdx.x * blockDim.x + threadIdx.x;
    if (i < n) {
        size_t row = i >> 10;
        float w = wband[row];
        float mag = reb[i], ph = imb[i];
        float mo = mag * w * matt[i];
        float po = ph * w + patt[i];
        reb[i] = mo * cosf(po);
        imb[i] = mo * sinf(po);
    }
}

// ---------------------------------------------------------------- inverse DFT (real part)
__global__ void k_idft(const float* __restrict__ cre, const float* __restrict__ cim,
                       const float* __restrict__ twC, const float* __restrict__ twS,
                       float* __restrict__ out) {
    const int tid = threadIdx.x, lane = tid & 31, wave = tid >> 5;
    const int bg = blockIdx.x, mtile = blockIdx.y;
    const int col = blockIdx.z * 128 + wave * 16 + (lane & 15);
    const int kk = (lane >> 4) * 2;
    const int m = mtile * 16 + (lane & 15);
    const float* Rb = cre + (size_t)bg * L_ * D_;
    const float* Ib = cim + (size_t)bg * L_ * D_;

    v8f acc = vzero8();
    for (int k = 0; k < L_; k += 4) {
        v2f ac = {0.f, 0.f}, as = {0.f, 0.f}, br, bi;
        if (m < L_) {
            ac = *(const v2f*)&twC[m * L_ + k + kk];
            as = *(const v2f*)&twS[m * L_ + k + kk];
        }
        br.x = Rb[(size_t)(k + kk) * D_ + col];
        br.y = Rb[(size_t)(k + kk + 1) * D_ + col];
        bi.x = Ib[(size_t)(k + kk) * D_ + col];
        bi.y = Ib[(size_t)(k + kk + 1) * D_ + col];
        acc = wmma_f32(ac, br, acc);     // + Re * cos
        acc = wmma_f32(as, bi, acc);     // + Im * sin (sign folded into twS)
    }
    int b = bg / G_, g = bg % G_;
    #pragma unroll
    for (int t = 0; t < 8; ++t) {
        int row = mtile * 16 + t + 8 * (lane >> 4);
        if (row < L_) {
            out[((size_t)b * SEQ_ + 1 + (size_t)g * L_ + row) * D_ + col] =
                acc[t] * (1.f / (float)L_);
        }
    }
}

// ---------------------------------------------------------------- cls passthrough
__global__ void k_cls(const float* __restrict__ x, float* __restrict__ out) {
    int i = blockIdx.x * blockDim.x + threadIdx.x;
    if (i < B_ * D_) {
        int b = i >> 10, d = i & (D_ - 1);
        out[(size_t)b * SEQ_ * D_ + d] = x[(size_t)b * SEQ_ * D_ + d];
    }
}

// ================================================================ launch
extern "C" void kernel_launch(void* const* d_in, const int* in_sizes, int n_in,
                              void* d_out, int out_size, void* d_ws, size_t ws_size,
                              hipStream_t stream) {
    (void)in_sizes; (void)n_in; (void)out_size; (void)ws_size;

    const float* x      = (const float*)d_in[0];
    const float* cg_w1  = (const float*)d_in[1];
    const float* cg_b1  = (const float*)d_in[2];
    const float* cg_lnw = (const float*)d_in[3];
    const float* cg_lnb = (const float*)d_in[4];
    const float* cg_w2  = (const float*)d_in[5];
    const float* cg_b2  = (const float*)d_in[6];
    const float* conv_w = (const float*)d_in[7];
    const float* conv_b = (const float*)d_in[8];
    const float* mag_w1 = (const float*)d_in[9];
    const float* mag_b1 = (const float*)d_in[10];
    const float* mag_lnw= (const float*)d_in[11];
    const float* mag_lnb= (const float*)d_in[12];
    const float* mag_w2 = (const float*)d_in[13];
    const float* mag_b2 = (const float*)d_in[14];
    const float* ph_w1  = (const float*)d_in[15];
    const float* ph_b1  = (const float*)d_in[16];
    const float* ph_lnw = (const float*)d_in[17];
    const float* ph_lnb = (const float*)d_in[18];
    const float* ph_w2  = (const float*)d_in[19];
    const float* ph_b2  = (const float*)d_in[20];
    const float* bd_w1  = (const float*)d_in[21];
    const float* bd_b1  = (const float*)d_in[22];
    const float* bd_lnw = (const float*)d_in[23];
    const float* bd_lnb = (const float*)d_in[24];
    const float* bd_w2  = (const float*)d_in[25];
    const float* bd_b2  = (const float*)d_in[26];
    float* out = (float*)d_out;

    const size_t SZ_GF  = (size_t)B_ * L_ * D_;
    const size_t SZ_BIG = (size_t)ROWS_ * D_;
    float* ws = (float*)d_ws;
    size_t o = 0;
    float* gf    = ws + o; o += SZ_GF;
    float* xa    = ws + o; o += SZ_BIG;
    float* reb   = ws + o; o += SZ_BIG;
    float* imb   = ws + o; o += SZ_BIG;
    float* matt  = ws + o; o += SZ_BIG;
    float* patt  = ws + o; o += SZ_BIG;
    float* wband = ws + o; o += (size_t)ROWS_;
    float* twC   = ws + o; o += (size_t)L_ * L_;
    float* twS   = ws + o; o += (size_t)L_ * L_;
    o = (o + 63) & ~(size_t)63;
    __bf16* bfb = (__bf16*)(ws + o);
    size_t ob = 0;
    __bf16* xg1b   = bfb + ob; ob += SZ_BIG;
    __bf16* magb   = bfb + ob; ob += SZ_BIG;
    __bf16* phb    = bfb + ob; ob += SZ_BIG;
    __bf16* cgw1t  = bfb + ob; ob += (size_t)512 * 2048;
    __bf16* convt  = bfb + ob; ob += (size_t)3 * D_ * D_;
    __bf16* magw1t = bfb + ob; ob += (size_t)256 * 1024;
    __bf16* magw2t = bfb + ob; ob += (size_t)1024 * 256;
    __bf16* phw1t  = bfb + ob; ob += (size_t)256 * 1024;
    __bf16* phw2t  = bfb + ob; ob += (size_t)1024 * 256;

    // ---- weight prep
    k_cvt_transpose<<<(2048 * 512 + 255) / 256, 256, 0, stream>>>(cg_w1, cgw1t, 2048, 512);
    k_cvt_transpose<<<(1024 * 256 + 255) / 256, 256, 0, stream>>>(mag_w1, magw1t, 1024, 256);
    k_cvt_transpose<<<(256 * 1024 + 255) / 256, 256, 0, stream>>>(mag_w2, magw2t, 256, 1024);
    k_cvt_transpose<<<(1024 * 256 + 255) / 256, 256, 0, stream>>>(ph_w1, phw1t, 1024, 256);
    k_cvt_transpose<<<(256 * 1024 + 255) / 256, 256, 0, stream>>>(ph_w2, phw2t, 256, 1024);
    k_cvt_conv<<<(3 * D_ * D_ + 255) / 256, 256, 0, stream>>>(conv_w, convt);
    k_twiddle<<<(L_ * L_ + 255) / 256, 256, 0, stream>>>(twC, twS);

    // ---- pipeline
    k_gf<<<(B_ * L_ * D_ / 4 + 255) / 256, 256, 0, stream>>>(x, gf);
    k_gate_xg<<<ROWS_ / 16, 256, 0, stream>>>(x, gf, cgw1t, cg_b1, cg_lnw, cg_lnb,
                                              cg_w2, cg_b2, xg1b);
    k_conv<<<dim3(ROWS_ / 16, 4), 256, 0, stream>>>(xg1b, convt, conv_b, xa);
    k_dft<<<dim3(B_ * G_, 13, 8), 256, 0, stream>>>(xa, twC, twS, reb, imb);
    k_band<<<B_ * G_, 256, 0, stream>>>(xa, bd_w1, bd_b1, bd_lnw, bd_lnb, bd_w2, bd_b2, wband);
    k_magph<<<(int)((SZ_BIG / 4 + 255) / 256), 256, 0, stream>>>(reb, imb, magb, phb, SZ_BIG / 4);
    k_mlp_att<<<ROWS_ / 16, 256, 0, stream>>>(magb, magw1t, mag_b1, mag_lnw, mag_lnb,
                                              magw2t, mag_b2, 0, matt);
    k_mlp_att<<<ROWS_ / 16, 256, 0, stream>>>(phb, phw1t, ph_b1, ph_lnw, ph_lnb,
                                              phw2t, ph_b2, 1, patt);
    k_combine<<<(int)((SZ_BIG + 255) / 256), 256, 0, stream>>>(reb, imb, matt, patt,
                                                               wband, SZ_BIG);
    k_idft<<<dim3(B_ * G_, 13, 8), 256, 0, stream>>>(reb, imb, twC, twS, out);
    k_cls<<<(B_ * D_ + 255) / 256, 256, 0, stream>>>(x, out);
}
